// TurboGNN_24721831756438
// MI455X (gfx1250) — matrix operand order repaired and checked by reading, not compiled, not run on hardware
//
#include <hip/hip_runtime.h>
#include <hip/hip_bf16.h>
#include <stdint.h>

#define Nn 50000
#define Ee 400000
#define Gg 512
#define Hh 256
#define Ll 5
#define Cc 6
#define EDd 7

typedef __attribute__((ext_vector_type(16))) __bf16 v16bf;
typedef __attribute__((ext_vector_type(8)))  float  v8f;
typedef __attribute__((ext_vector_type(4)))  unsigned int v4u;
typedef __attribute__((ext_vector_type(8)))  int v8i;
typedef __attribute__((ext_vector_type(4)))  int v4i;

static __device__ __forceinline__ float atomicMaxFloat(float* addr, float value) {
    if (value >= 0.0f)
        return __int_as_float(atomicMax((int*)addr, __float_as_int(value)));
    else
        return __uint_as_float(atomicMin((unsigned int*)addr, __float_as_uint(value)));
}

// ---------------------------------------------------------------------------
// TDM 2D tile load: [tileRows x 32] bf16 tile, row stride = strideK elements,
// into LDS with pad_enable (16 dwords data + 4 dwords pad = 40-element rows).
// dim0rem/dim1rem: remaining tensor extent from tile start (HW zero-fills OOB).
// D# bit layout per CDNA5 ISA 8.3/8.4.
// ---------------------------------------------------------------------------
static __device__ __forceinline__ void tdm_load_2d(const __bf16* gptr, unsigned int ldsoff,
                                                   int strideK, int dim0rem, int dim1rem,
                                                   int tileRows) {
    unsigned long long ga = (unsigned long long)(uintptr_t)gptr;
    v4u g0;
    g0[0] = 1u;                                                  // count=1, user D#
    g0[1] = ldsoff;                                              // lds_addr
    g0[2] = (unsigned int)ga;                                    // global_addr[31:0]
    g0[3] = (unsigned int)((ga >> 32) & 0x1FFFFFFu) | (2u << 30); // [56:32] | type=2
    v8i g1;
    // wg_mask=0 | data_size=1(2B)<<16 | pad_enable<<20 | pad_interval=3(16dw)<<22
    // | pad_amount=3(4dw)<<25
    g1[0] = (int)((1u << 16) | (1u << 20) | (3u << 22) | (3u << 25));
    g1[1] = (int)(((unsigned int)dim0rem & 0xFFFFu) << 16);      // tensor_dim0[15:0]
    g1[2] = (int)(((unsigned int)dim0rem >> 16) |
                  (((unsigned int)dim1rem & 0xFFFFu) << 16));    // dim0 hi | dim1 lo
    g1[3] = (int)(((unsigned int)dim1rem >> 16) | (32u << 16));  // dim1 hi | tile_dim0=32
    g1[4] = tileRows;                                            // tile_dim1 | tile_dim2=0
    g1[5] = strideK;                                             // tensor_dim0_stride lo32
    g1[6] = 0;
    g1[7] = 0;
    v4i g2 = {0, 0, 0, 0};
    v4i g3 = {0, 0, 0, 0};
#if defined(__clang_major__) && (__clang_major__ >= 23)
    v8i g4 = {0, 0, 0, 0, 0, 0, 0, 0};
    __builtin_amdgcn_tensor_load_to_lds(g0, g1, g2, g3, g4, 0);
#else
    __builtin_amdgcn_tensor_load_to_lds(g0, g1, g2, g3, 0);
#endif
}

// ---------------------------------------------------------------------------
// bf16-WMMA GEMM:  C[M,N] = act(A[M,K](bf16) @ Bt[N,K](bf16, pre-transposed) + bias)
// Both A and B tiles staged by the Tensor Data Mover, double-buffered in LDS:
// wave 0 issues the next (A,B) TDM pair, then s_wait_tensorcnt(2) so the newer
// pair stays in flight while the older completes (TDM ops retire in order).
// Tile: BM=64, BN=128, BK=32; 8 waves = 2(M) x 4(N); wave tile 32x32.
// Requires K % 32 == 0.
// ---------------------------------------------------------------------------
#define GBM 64
#define GBN 128
#define GBK 32
#define LDSTR 40   // bf16 elements per LDS row: 32 data + 8 pad (= 16+4 dwords)

__global__ __launch_bounds__(256)
void gemm_bf16_wmma(const __bf16* __restrict__ A, const __bf16* __restrict__ Bt,
                    const float* __restrict__ bias, void* __restrict__ Cout,
                    int M, int N, int K, int relu, int outBf16)
{
    __shared__ __align__(16) __bf16 As[2][GBM * LDSTR];
    __shared__ __align__(16) __bf16 Bs[2][GBN * LDSTR];

    const int tid  = threadIdx.x;
    const int lane = tid & 31;
    const int w    = tid >> 5;
    const int wm   = w & 1;       // wave row: 32 rows each
    const int wn   = w >> 1;      // wave col 0..3: 32 cols each
    const int half = lane >> 4;
    const int l16  = lane & 15;
    const int m0   = blockIdx.x * GBM;
    const int n0   = blockIdx.y * GBN;
    const int nk   = K / GBK;

    const __bf16* Atile = A  + (size_t)m0 * K;
    const __bf16* Btile = Bt + (size_t)n0 * K;

    v8f acc[2][2] = {};

    if (w == 0) {   // prologue: stage k-step 0 into buffer 0
        tdm_load_2d(Atile, (unsigned int)(uintptr_t)&As[0][0], K, K, M - m0, GBM);
        tdm_load_2d(Btile, (unsigned int)(uintptr_t)&Bs[0][0], K, K, N - n0, GBN);
    }

    for (int it = 0; it < nk; ++it) {
        if (w == 0) {
            if (it + 1 < nk) {
                int k1 = (it + 1) * GBK;
                int b1 = (it + 1) & 1;
                tdm_load_2d(Atile + k1, (unsigned int)(uintptr_t)&As[b1][0],
                            K, K - k1, M - m0, GBM);
                tdm_load_2d(Btile + k1, (unsigned int)(uintptr_t)&Bs[b1][0],
                            K, K - k1, N - n0, GBN);
                __builtin_amdgcn_s_wait_tensorcnt(2);  // current pair complete
            } else {
                __builtin_amdgcn_s_wait_tensorcnt(0);
            }
        }
        __syncthreads();   // tiles for step `it` visible to all waves

        const __bf16* Ab = &As[it & 1][0];
        const __bf16* Bb = &Bs[it & 1][0];

        // ---- A fragments (16x32 bf16 layout, ISA 7.12.2) ----
        v16bf af[2];
        #pragma unroll
        for (int mt = 0; mt < 2; ++mt) {
            const __bf16* ar = &Ab[(wm * 32 + mt * 16 + l16) * LDSTR];
            const int kb = half * 8;
            #pragma unroll
            for (int j = 0; j < 8; ++j) {
                af[mt][j]     = ar[kb + j];
                af[mt][8 + j] = ar[16 + kb + j];
            }
        }
        // ---- B fragments + wmma ----
        #pragma unroll
        for (int nt = 0; nt < 2; ++nt) {
            v16bf bfr;
            const __bf16* br = &Bb[(wn * 32 + nt * 16 + l16) * LDSTR];
            #pragma unroll
            for (int i = 0; i < 16; ++i) bfr[i] = br[(i < 8) ? (half * 8 + i) : (16 + half * 8 + (i - 8))];
            acc[0][nt] = __builtin_amdgcn_wmma_f32_16x16x32_bf16(
                false, af[0], false, bfr, (short)0, acc[0][nt], false, false);
            acc[1][nt] = __builtin_amdgcn_wmma_f32_16x16x32_bf16(
                false, af[1], false, bfr, (short)0, acc[1][nt], false, false);
        }
        __syncthreads();   // reads done before wave0 overwrites this buffer (it+2)
    }

    // ---- epilogue: C/D layout row = half*8 + r, col = l16 ----
    #pragma unroll
    for (int mt = 0; mt < 2; ++mt) {
        #pragma unroll
        for (int nt = 0; nt < 2; ++nt) {
            int col = n0 + wn * 32 + nt * 16 + l16;
            if (col >= N) continue;
            float bv = bias ? bias[col] : 0.0f;
            #pragma unroll
            for (int r = 0; r < 8; ++r) {
                int row = m0 + wm * 32 + mt * 16 + half * 8 + r;
                if (row < M) {
                    float v = acc[mt][nt][r] + bv;
                    if (relu) v = fmaxf(v, 0.0f);
                    if (outBf16)
                        ((__bf16*)Cout)[(size_t)row * N + col] = (__bf16)v;
                    else
                        ((float*)Cout)[(size_t)row * N + col] = v;
                }
            }
        }
    }
}

// ---------------------------------------------------------------------------
// Elementwise / scatter kernels
// ---------------------------------------------------------------------------
__global__ void fill_k(float* p, float v, long long n) {
    long long i = (long long)blockIdx.x * blockDim.x + threadIdx.x;
    long long stride = (long long)gridDim.x * blockDim.x;
    for (; i < n; i += stride) p[i] = v;
}

__global__ void cvt_bf16_k(const float* __restrict__ in, __bf16* __restrict__ out,
                           long long n) {
    long long i = (long long)blockIdx.x * blockDim.x + threadIdx.x;
    long long stride = (long long)gridDim.x * blockDim.x;
    for (; i < n; i += stride) out[i] = (__bf16)in[i];
}

// Wt[n*K+k] = (bf16)W[k*N+n]  (weights: one-time transpose+convert)
__global__ void transpose_cvt_k(const float* __restrict__ W, __bf16* __restrict__ Wt,
                                int K, int N) {
    int idx = blockIdx.x * blockDim.x + threadIdx.x;
    if (idx >= K * N) return;
    int k = idx / N, n = idx % N;
    Wt[(size_t)n * K + k] = (__bf16)W[idx];
}

__global__ void init_x_k(const int* __restrict__ x_ids, const float* __restrict__ emb,
                         float* __restrict__ x) {
    long long idx = (long long)blockIdx.x * blockDim.x + threadIdx.x;
    if (idx >= (long long)Nn * Hh) return;
    int i = (int)(idx >> 8), j = (int)(idx & 255);
    x[idx] = emb[(size_t)x_ids[i] * Hh + j];
}

__global__ void count_k(const int* __restrict__ batch, float* __restrict__ cnt) {
    int i = blockIdx.x * blockDim.x + threadIdx.x;
    if (i < Nn) atomicAdd(&cnt[batch[i]], 1.0f);
}

__global__ void invc_k(const float* __restrict__ cnt, float* __restrict__ invc) {
    int g = blockIdx.x * blockDim.x + threadIdx.x;
    if (g < Gg) invc[g] = 1.0f / fmaxf(cnt[g], 1.0f);
}

// hidden = relu(edge_attr @ ee_W1 + ee_b1) -> bf16. K=7 scalar, 16 edges/block.
__global__ __launch_bounds__(256)
void edge_hidden_k(const float* __restrict__ ea, const float* __restrict__ W1,
                   const float* __restrict__ b1, __bf16* __restrict__ hid) {
    __shared__ float sW[EDd * Hh];
    __shared__ float sb[Hh];
    const int tid = threadIdx.x;
    for (int i = tid; i < EDd * Hh; i += 256) sW[i] = W1[i];
    sb[tid] = b1[tid];
    __syncthreads();
    const int ebase = blockIdx.x * 16;
    for (int t = 0; t < 16; ++t) {
        int e = ebase + t;
        if (e >= Ee) break;
        const float* ar = ea + (size_t)e * EDd;
        float s = sb[tid];
        #pragma unroll
        for (int k = 0; k < EDd; ++k) s += ar[k] * sW[k * Hh + tid];
        hid[(size_t)e * Hh + tid] = (__bf16)fmaxf(s, 0.0f);
    }
}

// msg = relu(x[src] + edge_emb); agg[dst] += msg   (4 floats per thread)
__global__ void msg_agg_k(const int* __restrict__ ei, const float* __restrict__ x,
                          const float* __restrict__ ee, float* __restrict__ agg) {
    long long idx = (long long)blockIdx.x * blockDim.x + threadIdx.x;
    if (idx >= (long long)Ee * (Hh / 4)) return;
    int e = (int)(idx >> 6);
    int j = (int)(idx & 63) << 2;
    int s = ei[e];
    int d = ei[Ee + e];
    float4 xv = *reinterpret_cast<const float4*>(x + (size_t)s * Hh + j);
    float4 ev = *reinterpret_cast<const float4*>(ee + (size_t)e * Hh + j);
    float* ap = agg + (size_t)d * Hh + j;
    atomicAdd(ap + 0, fmaxf(xv.x + ev.x, 0.0f));
    atomicAdd(ap + 1, fmaxf(xv.y + ev.y, 0.0f));
    atomicAdd(ap + 2, fmaxf(xv.z + ev.z, 0.0f));
    atomicAdd(ap + 3, fmaxf(xv.w + ev.w, 0.0f));
}

// hin(bf16) = (1+eps[l])*x + agg
__global__ void hin_k(const float* __restrict__ x, const float* __restrict__ agg,
                      const float* __restrict__ eps, int l, __bf16* __restrict__ hin) {
    long long idx = (long long)blockIdx.x * blockDim.x + threadIdx.x;
    if (idx >= (long long)Nn * Hh) return;
    hin[idx] = (__bf16)((1.0f + eps[l]) * x[idx] + agg[idx]);
}

__global__ void gsum_k(const float* __restrict__ h, const int* __restrict__ batch,
                       float* __restrict__ gsum) {
    long long idx = (long long)blockIdx.x * blockDim.x + threadIdx.x;
    if (idx >= (long long)Nn * Hh) return;
    int i = (int)(idx >> 8), j = (int)(idx & 255);
    atomicAdd(&gsum[(size_t)batch[i] * Hh + j], h[idx]);
}

__global__ void cenvar_k(const float* __restrict__ h, const int* __restrict__ batch,
                         const float* __restrict__ gsum, const float* __restrict__ invc,
                         const float* __restrict__ scale, float* __restrict__ cen,
                         float* __restrict__ gvar) {
    long long idx = (long long)blockIdx.x * blockDim.x + threadIdx.x;
    if (idx >= (long long)Nn * Hh) return;
    int i = (int)(idx >> 8), j = (int)(idx & 255);
    int g = batch[i];
    float mean = gsum[(size_t)g * Hh + j] * invc[g];
    float c = h[idx] - scale[j] * mean;
    cen[idx] = c;
    atomicAdd(&gvar[(size_t)g * Hh + j], c * c);
}

__global__ void norm_k(const float* __restrict__ cen, const int* __restrict__ batch,
                       const float* __restrict__ gvar, const float* __restrict__ invc,
                       const float* __restrict__ gw, const float* __restrict__ gb,
                       float* __restrict__ x, float* __restrict__ feat) {
    long long idx = (long long)blockIdx.x * blockDim.x + threadIdx.x;
    if (idx >= (long long)Nn * Hh) return;
    int i = (int)(idx >> 8), j = (int)(idx & 255);
    int g = batch[i];
    float var = gvar[(size_t)g * Hh + j] * invc[g];
    float hn  = gw[j] * cen[idx] * rsqrtf(var + 1e-5f) + gb[j];
    float xn  = fmaxf(hn, 0.0f) + x[idx];
    x[idx] = xn;
    feat[idx] += xn;
}

__global__ void xmean_k(const float* __restrict__ feat, float* __restrict__ xm) {
    long long idx = (long long)blockIdx.x * blockDim.x + threadIdx.x;
    if (idx >= (long long)Nn * Hh) return;
    xm[idx] = feat[idx] * (1.0f / (float)Ll);
}

__global__ void rep_init_k(float* __restrict__ rep) {
    int idx = blockIdx.x * blockDim.x + threadIdx.x;
    if (idx >= Gg * 2 * Hh) return;
    rep[idx] = ((idx & 511) < Hh) ? 0.0f : __int_as_float(0xff800000);
}

// one block (256 thr) per node: att = sigmoid(xm . att_W + b); scatter wsum & max
__global__ __launch_bounds__(256)
void att_pool_k(const float* __restrict__ xm, const int* __restrict__ batch,
                const float* __restrict__ attW, const float* __restrict__ attb,
                float* __restrict__ rep) {
    __shared__ float red[256];
    const int i = blockIdx.x;
    const int j = threadIdx.x;
    float xv = xm[(size_t)i * Hh + j];
    red[j] = xv * attW[j];
    __syncthreads();
    for (int s = 128; s > 0; s >>= 1) {
        if (j < s) red[j] += red[j + s];
        __syncthreads();
    }
    float att = 1.0f / (1.0f + __expf(-(red[0] + attb[0])));
    int g = batch[i];
    atomicAdd(rep + (size_t)g * 2 * Hh + j, xv * att);
    atomicMaxFloat(rep + (size_t)g * 2 * Hh + Hh + j, xv);
}

// out[g,c] = cmid[g,:] . cls_W2[:,c] + b2[c]
__global__ void cls_out_k(const float* __restrict__ cmid, const float* __restrict__ W2,
                          const float* __restrict__ b2, float* __restrict__ out) {
    int idx = blockIdx.x * blockDim.x + threadIdx.x;
    if (idx >= Gg * Cc) return;
    int g = idx / Cc, c = idx % Cc;
    float s = b2[c];
    for (int j = 0; j < Hh; ++j) s += cmid[(size_t)g * Hh + j] * W2[j * Cc + c];
    out[idx] = s;
}

// ---------------------------------------------------------------------------
extern "C" void kernel_launch(void* const* d_in, const int* in_sizes, int n_in,
                              void* d_out, int out_size, void* d_ws, size_t ws_size,
                              hipStream_t stream) {
    const int*   x_ids      = (const int*)  d_in[0];
    const int*   edge_index = (const int*)  d_in[1];
    const int*   batch      = (const int*)  d_in[2];
    const float* edge_attr  = (const float*)d_in[3];
    const float* emb_table  = (const float*)d_in[4];
    const float* ee_W1      = (const float*)d_in[5];
    const float* ee_b1      = (const float*)d_in[6];
    const float* ee_W2      = (const float*)d_in[7];
    const float* ee_b2      = (const float*)d_in[8];
    const float* eps        = (const float*)d_in[9];
    const float* conv_W1    = (const float*)d_in[10];
    const float* conv_b1    = (const float*)d_in[11];
    const float* conv_W2    = (const float*)d_in[12];
    const float* conv_b2    = (const float*)d_in[13];
    const float* gn_weight  = (const float*)d_in[14];
    const float* gn_bias    = (const float*)d_in[15];
    const float* gn_scale   = (const float*)d_in[16];
    const float* att_W      = (const float*)d_in[17];
    const float* att_b      = (const float*)d_in[18];
    const float* cls_W1     = (const float*)d_in[19];
    const float* cls_b1     = (const float*)d_in[20];
    const float* cls_W2     = (const float*)d_in[21];
    const float* cls_b2     = (const float*)d_in[22];
    float* out = (float*)d_out;

    // ---- workspace layout ----
    float* p = (float*)d_ws;
    float* edge_emb = p; p += (size_t)Ee * Hh;       // [E,H] f32, live all layers
    float* x        = p; p += (size_t)Nn * Hh;
    float* agg      = p; p += (size_t)Nn * Hh;
    float* tmp      = p; p += (size_t)Nn * Hh;       // cen / xm (f32)
    float* hbuf     = p; p += (size_t)Nn * Hh;
    float* feat     = p; p += (size_t)Nn * Hh;
    float* gsum     = p; p += (size_t)Gg * Hh;
    float* gvar     = p; p += (size_t)Gg * Hh;
    float* cnt      = p; p += Gg;
    float* invc     = p; p += Gg;
    float* rep      = p; p += (size_t)Gg * 2 * Hh;
    float* cmid     = p; p += (size_t)Gg * Hh;
    // bf16 region
    __bf16* bp      = (__bf16*)p;
    __bf16* ehid_bf = bp; bp += (size_t)Ee * Hh;     // edge MLP hidden (GEMM A)
    __bf16* hin_bf  = bp; bp += (size_t)Nn * Hh;     // GIN MLP input   (GEMM A)
    __bf16* hmid_bf = bp; bp += (size_t)Nn * 2 * Hh; // GIN MLP hidden  (GEMM A)
    __bf16* rep_bf  = bp; bp += (size_t)Gg * 2 * Hh; // classifier in   (GEMM A)
    __bf16* wt_ee   = bp; bp += (size_t)Hh * Hh;           // ee_W2^T   [256][256]
    __bf16* wt_c1   = bp; bp += (size_t)Ll * Hh * 2 * Hh;  // conv_W1^T [512][256] x L
    __bf16* wt_c2   = bp; bp += (size_t)Ll * 2 * Hh * Hh;  // conv_W2^T [256][512] x L
    __bf16* wt_cls  = bp; bp += (size_t)2 * Hh * Hh;       // cls_W1^T  [256][512]

    const long long NH = (long long)Nn * Hh;
    const int nhBlocks = (int)((NH + 255) / 256);

    // ---- one-time weight transpose+convert (tiny) ----
    transpose_cvt_k<<<(Hh * Hh + 255) / 256, 256, 0, stream>>>(ee_W2, wt_ee, Hh, Hh);
    for (int l = 0; l < Ll; ++l) {
        transpose_cvt_k<<<(Hh * 2 * Hh + 255) / 256, 256, 0, stream>>>(
            conv_W1 + (size_t)l * Hh * 2 * Hh, wt_c1 + (size_t)l * Hh * 2 * Hh, Hh, 2 * Hh);
        transpose_cvt_k<<<(2 * Hh * Hh + 255) / 256, 256, 0, stream>>>(
            conv_W2 + (size_t)l * 2 * Hh * Hh, wt_c2 + (size_t)l * 2 * Hh * Hh, 2 * Hh, Hh);
    }
    transpose_cvt_k<<<(2 * Hh * Hh + 255) / 256, 256, 0, stream>>>(cls_W1, wt_cls, 2 * Hh, Hh);

    // ---- init ----
    fill_k<<<4096, 256, 0, stream>>>(feat, 0.0f, NH);
    fill_k<<<4, 256, 0, stream>>>(cnt, 0.0f, Gg);
    init_x_k<<<nhBlocks, 256, 0, stream>>>(x_ids, emb_table, x);
    count_k<<<(Nn + 255) / 256, 256, 0, stream>>>(batch, cnt);
    invc_k<<<(Gg + 255) / 256, 256, 0, stream>>>(cnt, invc);

    // ---- edge MLP ----
    edge_hidden_k<<<(Ee + 15) / 16, 256, 0, stream>>>(edge_attr, ee_W1, ee_b1, ehid_bf);
    {
        dim3 grid((Ee + GBM - 1) / GBM, Hh / GBN);   // 6250 x 2
        gemm_bf16_wmma<<<grid, 256, 0, stream>>>(ehid_bf, wt_ee, ee_b2, edge_emb,
                                                 Ee, Hh, Hh, 0, 0);
    }

    // ---- GINE layers ----
    for (int l = 0; l < Ll; ++l) {
        fill_k<<<4096, 256, 0, stream>>>(agg, 0.0f, NH);
        msg_agg_k<<<(int)(((long long)Ee * (Hh / 4) + 255) / 256), 256, 0, stream>>>(
            edge_index, x, edge_emb, agg);
        hin_k<<<nhBlocks, 256, 0, stream>>>(x, agg, eps, l, hin_bf);
        {
            dim3 g1((Nn + GBM - 1) / GBM, (2 * Hh) / GBN);  // 782 x 4
            gemm_bf16_wmma<<<g1, 256, 0, stream>>>(hin_bf,
                wt_c1 + (size_t)l * Hh * 2 * Hh, conv_b1 + (size_t)l * 2 * Hh,
                hmid_bf, Nn, 2 * Hh, Hh, 1, 1);
            dim3 g2((Nn + GBM - 1) / GBM, Hh / GBN);        // 782 x 2
            gemm_bf16_wmma<<<g2, 256, 0, stream>>>(hmid_bf,
                wt_c2 + (size_t)l * 2 * Hh * Hh, conv_b2 + (size_t)l * Hh,
                hbuf, Nn, Hh, 2 * Hh, 0, 0);
        }
        fill_k<<<512, 256, 0, stream>>>(gsum, 0.0f, (long long)Gg * Hh);
        fill_k<<<512, 256, 0, stream>>>(gvar, 0.0f, (long long)Gg * Hh);
        gsum_k<<<nhBlocks, 256, 0, stream>>>(hbuf, batch, gsum);
        cenvar_k<<<nhBlocks, 256, 0, stream>>>(hbuf, batch, gsum, invc,
                                               gn_scale + (size_t)l * Hh, tmp, gvar);
        norm_k<<<nhBlocks, 256, 0, stream>>>(tmp, batch, gvar, invc,
                                             gn_weight + (size_t)l * Hh,
                                             gn_bias + (size_t)l * Hh, x, feat);
    }

    // ---- readout ----
    xmean_k<<<nhBlocks, 256, 0, stream>>>(feat, tmp);             // tmp = xm
    rep_init_k<<<(Gg * 2 * Hh + 255) / 256, 256, 0, stream>>>(rep);
    att_pool_k<<<Nn, 256, 0, stream>>>(tmp, batch, att_W, att_b, rep);
    cvt_bf16_k<<<256, 256, 0, stream>>>(rep, rep_bf, (long long)Gg * 2 * Hh);
    {
        dim3 gc((Gg + GBM - 1) / GBM, Hh / GBN);                  // 8 x 2
        gemm_bf16_wmma<<<gc, 256, 0, stream>>>(rep_bf, wt_cls, cls_b1, cmid,
                                               Gg, Hh, 2 * Hh, 1, 0);
    }
    cls_out_k<<<(Gg * Cc + 255) / 256, 256, 0, stream>>>(cmid, cls_W2, cls_b2, out);
}
